// Attention_logits_12979391169263
// MI455X (gfx1250) — matrix operand order repaired and checked by reading, not compiled, not run on hardware
//
#include <hip/hip_runtime.h>

// MI455X (gfx1250) axial attention: bf16 WMMA + async global->LDS staging.
// All matrix math through v_wmma_f32_16x16x32_bf16 (wave32); shared B-operands
// staged in LDS (320KB/WGP) via GLOBAL_LOAD_ASYNC_TO_LDS_B128 when available.

typedef __bf16 bf16_t;
typedef __attribute__((ext_vector_type(16))) __bf16 v16bf;
typedef __attribute__((ext_vector_type(8)))  __bf16 v8bf;
typedef __attribute__((ext_vector_type(8)))  float  v8f;
typedef int v4i __attribute__((vector_size(16)));   // int4, matches builtin param

#if defined(__AMDGCN__) && defined(__has_builtin)
#if __has_builtin(__builtin_amdgcn_global_load_async_to_lds_b128) && \
    __has_builtin(__builtin_amdgcn_s_wait_asynccnt)
#define HAVE_ASYNC 1
#endif
#endif
#ifndef HAVE_ASYNC
#define HAVE_ASYNC 0
#endif

#if HAVE_ASYNC
typedef __attribute__((address_space(1))) v4i glb_v4i_t;   // global int4
typedef __attribute__((address_space(3))) v4i lds_v4i_t;   // LDS int4
#endif

// Problem constants: B=4, N=256, M=256, C=256
#define GROUPS   1024          // B*256 attention groups per pass
#define ROWS_ALL (1024 * 256)  // total (group,pos) rows per pass
#define NMC      16777216UL    // N*M*C
#define MC_      65536UL       // M*C
#define NM_      65536UL       // N*M
#define PLD      264           // LDS row stride (bf16): 528B = 16B-aligned, 132 dwords
                               // (132 % 64 = 4 -> conflict-free B-fragment reads)

// ---------------------------------------------------------------------------
// W [C,3C] f32 (row-major, x@W) -> Wt [3C,C] bf16 (B-columns become K-contiguous)
__global__ void __launch_bounds__(256) wt_convert_kernel(const float* __restrict__ W,
                                                         bf16_t* __restrict__ Wt) {
    int idx = blockIdx.x * 256 + threadIdx.x;   // 768*256 threads
    int j = idx >> 8;                           // output feature 0..767
    int c = idx & 255;                          // input channel
    Wt[(size_t)j * 256 + c] = (bf16_t)W[(size_t)c * 768 + j];
}

// ---------------------------------------------------------------------------
// Stage a 256x256 bf16 panel (row stride 256) into LDS (row stride PLD).
// Async path: 8192 b128 transfers tracked by ASYNCcnt; fallback: plain copies.
__device__ __forceinline__ void stage_panel_256(const bf16_t* __restrict__ src,
                                                bf16_t* dst, int t) {
#pragma unroll
    for (int i = 0; i < 32; ++i) {
        const int chunk = t + (i << 8);        // 0..8191
        const int row   = chunk >> 5;          // 32 x 16B per 512B row
        const int c16   = (chunk & 31) << 3;   // element offset (8 bf16 = 16B)
#if HAVE_ASYNC
        __builtin_amdgcn_global_load_async_to_lds_b128(
            (glb_v4i_t*)(src + (size_t)row * 256 + c16),
            (lds_v4i_t*)(dst + row * PLD + c16), 0, 0);
#else
        *(v8bf*)(dst + row * PLD + c16) = *(const v8bf*)(src + (size_t)row * 256 + c16);
#endif
    }
}

__device__ __forceinline__ void wait_async_all() {
#if HAVE_ASYNC
    __builtin_amdgcn_s_wait_asynccnt(0);
#endif
}

// ---------------------------------------------------------------------------
// A-fragment (16-bit, 16x32) from row-major bf16 (works for global or LDS).
// lanes 0-15: K = k0+0..15 for row r0+lane; lanes 16-31: +16.
__device__ __forceinline__ v16bf load_a_bf16(const bf16_t* __restrict__ base, int ld,
                                             int r0, int k0, int lane) {
    const bf16_t* p = base + (size_t)(r0 + (lane & 15)) * ld + k0 + ((lane >> 4) << 4);
    v8bf lo = *(const v8bf*)p;
    v8bf hi = *(const v8bf*)(p + 8);
    v16bf a;
#pragma unroll
    for (int i = 0; i < 8; ++i) { a[i] = lo[i]; a[i + 8] = hi[i]; }
    return a;
}

// ---------------------------------------------------------------------------
// QKV projection: X rows (fp32, strided) x Wt -> Q,K (bf16 [G*L][256]) and
// V transposed (bf16 Vt[g][c][m]). One wave per 16-row tile; A fragments
// cached in VGPRs; Wt staged chunk-by-chunk (Q/K/V blocks) into LDS and
// shared by all 8 waves (8x cut in Wt L2 traffic).
__global__ void __launch_bounds__(256) qkv_wmma_kernel(
    const float* __restrict__ X, const float* __restrict__ bias,
    const bf16_t* __restrict__ Wt,
    bf16_t* __restrict__ Qb, bf16_t* __restrict__ Kb, bf16_t* __restrict__ Vt,
    size_t xs0, size_t xs1, size_t xs2)  // offs: (g/256)*xs0 + (g%256)*xs1 + l*xs2 + c
{
    extern __shared__ char smem[];
    bf16_t* Wst = (bf16_t*)smem;                 // 256 x PLD bf16 staging panel

    const int t = threadIdx.x;
    const int lane = t & 31;
    const int wave = t >> 5;
    const int rowTile = blockIdx.x * 8 + wave;   // 16384 tiles of 16 rows
    const int r0 = rowTile << 4;
    const int g  = r0 >> 8;                      // all 16 rows share one group
    const size_t gq = (size_t)g << 16;           // g * 65536

    const int row = r0 + (lane & 15);
    const float* rowp = X + (size_t)(g >> 8) * xs0 + (size_t)(g & 255) * xs1
                          + (size_t)(row & 255) * xs2;
    const int hh8 = (lane >> 4) << 3;            // accumulator row half (D rows hh8+v)
    const int kb  = (lane >> 4) << 4;            // B-frag K half
    const int nlane = lane & 15;

    // Cache 8 A fragments covering K = 0..255 (f32 -> bf16 conversion here).
    v16bf afr[8];
#pragma unroll
    for (int kt = 0; kt < 8; ++kt) {
        const float* p = rowp + (kt << 5) + kb;
        v16bf a;
#pragma unroll
        for (int i = 0; i < 8; ++i) { a[i] = (bf16_t)p[i]; a[i + 8] = (bf16_t)p[i + 8]; }
        afr[kt] = a;
    }

#pragma unroll 1
    for (int chunk = 0; chunk < 3; ++chunk) {    // 0: Q cols, 1: K cols, 2: V cols
        stage_panel_256(Wt + (size_t)chunk * 256 * 256, Wst, t);
        wait_async_all();
        __syncthreads();

#pragma unroll 1
        for (int ct = 0; ct < 16; ++ct) {
            const int n0 = ct << 4;
            const bf16_t* bp = Wst + (size_t)(n0 + nlane) * PLD + kb;
            v8f acc = {};
#pragma unroll
            for (int kt = 0; kt < 8; ++kt) {
                v16bf b = *(const v16bf*)(bp + (kt << 5));
                acc = __builtin_amdgcn_wmma_f32_16x16x32_bf16(false, afr[kt], false, b,
                                                              (short)0, acc, false, false);
            }
            const int col = (chunk << 8) + n0 + nlane;
            const float bv = bias[col];
            if (chunk == 0) {                    // Q, row-major bf16
#pragma unroll
                for (int v = 0; v < 8; ++v)
                    Qb[(size_t)(r0 + v + hh8) * 256 + (col)] = (bf16_t)(acc[v] + bv);
            } else if (chunk == 1) {             // K, row-major bf16
#pragma unroll
                for (int v = 0; v < 8; ++v)
                    Kb[(size_t)(r0 + v + hh8) * 256 + (col - 256)] = (bf16_t)(acc[v] + bv);
            } else {                             // V, stored transposed: Vt[g][c][m]
                const int cv = col - 512;
                v8bf pk;
#pragma unroll
                for (int v = 0; v < 8; ++v) pk[v] = (bf16_t)(acc[v] + bv);
                *(v8bf*)(Vt + gq + (size_t)cv * 256 + (r0 & 255) + hh8) = pk;  // 16B store
            }
        }
        __syncthreads();                          // panel consumed before restaging
    }
}

// ---------------------------------------------------------------------------
// Per-group attention: S = mask(Q K^T * scale) -> softmax (LDS) -> out = P V.
// LDS: prob tile (132KB) + K/V staging panel (132KB) + mask vec; K and V are
// each pulled from memory exactly once per group (V staged async under softmax).
__global__ void __launch_bounds__(256) axial_attn_kernel(
    const bf16_t* __restrict__ Qb, const bf16_t* __restrict__ Kb,
    const bf16_t* __restrict__ Vt,
    const float* __restrict__ mask, float* __restrict__ OUT,
    size_t ms0, size_t ms1, size_t ms2,
    size_t os0, size_t os1, size_t os2)
{
    extern __shared__ char smem[];
    bf16_t* P    = (bf16_t*)smem;                        // 256 * PLD bf16
    bf16_t* KVst = (bf16_t*)(smem + 256 * PLD * 2);      // 256 * PLD bf16
    float*  mvec = (float*)(smem + 2 * 256 * PLD * 2);   // 256 f32

    const int g = blockIdx.x;
    const size_t gq = (size_t)g << 16;
    const int t = threadIdx.x;
    const int lane = t & 31, wave = t >> 5;
    const int nlane = lane & 15;
    const int hh8 = (lane >> 4) << 3;
    const int kb  = (lane >> 4) << 4;

    mvec[t] = mask[(size_t)(g >> 8) * ms0 + (size_t)(g & 255) * ms1 + (size_t)t * ms2];
    stage_panel_256(Kb + gq, KVst, t);                   // K panel -> LDS
    wait_async_all();
    __syncthreads();

    // ---- Phase 1: masked scores into LDS (bf16) ----
    const bf16_t* Qg = Qb + gq;
#pragma unroll 1
    for (int rti = 0; rti < 2; ++rti) {
        const int r0 = (wave * 2 + rti) << 4;
        v16bf afr[8];
#pragma unroll
        for (int kt = 0; kt < 8; ++kt) afr[kt] = load_a_bf16(Qg, 256, r0, kt << 5, lane);
        float mrow[8];
#pragma unroll
        for (int v = 0; v < 8; ++v) mrow[v] = mvec[r0 + v + hh8];
#pragma unroll 1
        for (int ct = 0; ct < 16; ++ct) {
            const int n0 = ct << 4;
            const bf16_t* bp = KVst + (size_t)(n0 + nlane) * PLD + kb;
            v8f acc = {};
#pragma unroll
            for (int kt = 0; kt < 8; ++kt) {
                v16bf b = *(const v16bf*)(bp + (kt << 5));
                acc = __builtin_amdgcn_wmma_f32_16x16x32_bf16(false, afr[kt], false, b,
                                                              (short)0, acc, false, false);
            }
            const int col = n0 + nlane;
            const float mc = mvec[col];
#pragma unroll
            for (int v = 0; v < 8; ++v) {
                float s = acc[v] * 0.0625f;                  // * C^-0.5
                if (mrow[v] * mc == 0.0f) s = -10000.0f;     // pair mask
                P[(r0 + v + hh8) * PLD + col] = (bf16_t)s;
            }
        }
    }
    __syncthreads();                                  // everyone done with K panel

    // Kick off V panel staging; overlaps with the softmax below (ASYNCcnt).
    stage_panel_256(Vt + gq, KVst, t);

    // ---- Phase 2: row softmax (one row per thread) ----
    {
        bf16_t* pr = P + t * PLD;
        float mx = -3.0e38f;
        for (int i = 0; i < 256; ++i) mx = fmaxf(mx, (float)pr[i]);
        float sum = 0.0f;
        for (int i = 0; i < 256; ++i) {
            float e = __expf((float)pr[i] - mx);
            sum += e;
            pr[i] = (bf16_t)e;
        }
        const float inv = 1.0f / sum;
        for (int i = 0; i < 256; ++i) pr[i] = (bf16_t)((float)pr[i] * inv);
    }
    wait_async_all();
    __syncthreads();

    // ---- Phase 3: out = P @ V (A from LDS P, B from LDS V panel) ----
    const size_t outg = (size_t)(g >> 8) * os0 + (size_t)(g & 255) * os1;
#pragma unroll 1
    for (int rti = 0; rti < 2; ++rti) {
        const int r0 = (wave * 2 + rti) << 4;
        v16bf afr[8];
#pragma unroll
        for (int kt = 0; kt < 8; ++kt) afr[kt] = load_a_bf16(P, PLD, r0, kt << 5, lane);
#pragma unroll 1
        for (int ct = 0; ct < 16; ++ct) {
            const int n0 = ct << 4;
            const bf16_t* bp = KVst + (size_t)(n0 + nlane) * PLD + kb;
            v8f acc = {};
#pragma unroll
            for (int kt = 0; kt < 8; ++kt) {
                v16bf b = *(const v16bf*)(bp + (kt << 5));
                acc = __builtin_amdgcn_wmma_f32_16x16x32_bf16(false, afr[kt], false, b,
                                                              (short)0, acc, false, false);
            }
            const int ch = n0 + nlane;
#pragma unroll
            for (int v = 0; v < 8; ++v)
                OUT[outg + (size_t)(r0 + v + hh8) * os2 + ch] = acc[v];
        }
    }
}

// ---------------------------------------------------------------------------
extern "C" void kernel_launch(void* const* d_in, const int* in_sizes, int n_in,
                              void* d_out, int out_size, void* d_ws, size_t ws_size,
                              hipStream_t stream) {
    (void)in_sizes; (void)n_in; (void)out_size; (void)ws_size;
    const float* x    = (const float*)d_in[0];
    const float* mask = (const float*)d_in[1];
    const float* W    = (const float*)d_in[2];
    const float* bq   = (const float*)d_in[3];
    float* out = (float*)d_out;

    // Workspace: Wt 384KB | Qb 128MB | Kb 128MB | Vt 128MB | Y1 256MB (~641MB)
    char* ws = (char*)d_ws;
    bf16_t* Wt = (bf16_t*)ws;                size_t off = 768 * 256 * 2;
    bf16_t* Qb = (bf16_t*)(ws + off);        off += (size_t)ROWS_ALL * 256 * 2;
    bf16_t* Kb = (bf16_t*)(ws + off);        off += (size_t)ROWS_ALL * 256 * 2;
    bf16_t* Vt = (bf16_t*)(ws + off);        off += (size_t)ROWS_ALL * 256 * 2;
    float*  Y1 = (float*)(ws + off);

    const size_t smemQKV  = 256 * PLD * 2;                    // one staging panel
    const size_t smemAttn = 2 * 256 * PLD * 2 + 256 * 4;      // P + panel + mask

    wt_convert_kernel<<<768, 256, 0, stream>>>(W, Wt);

    // Pass 1: attend along M. groups g=(b,n): X strides (NMC, MC, C); mask (NM,256,1)
    qkv_wmma_kernel<<<2048, 256, smemQKV, stream>>>(x, bq, Wt, Qb, Kb, Vt,
                                                    NMC, MC_, 256);
    axial_attn_kernel<<<GROUPS, 256, smemAttn, stream>>>(Qb, Kb, Vt, mask, Y1,
                                                         NM_, 256, 1,
                                                         NMC, MC_, 256);

    // Pass 2: attend along N. groups g=(b,m): read Y1 with (NMC, C, MC);
    // mask (NM, 1, 256); write d_out in [b,n,m,c] layout via (NMC, C, MC).
    qkv_wmma_kernel<<<2048, 256, smemQKV, stream>>>(Y1, bq, Wt, Qb, Kb, Vt,
                                                    NMC, 256, MC_);
    axial_attn_kernel<<<GROUPS, 256, smemAttn, stream>>>(Qb, Kb, Vt, mask, out,
                                                         NM_, 1, 256,
                                                         NMC, 256, MC_);
}
